// ConsMaxAttention_55095840473610
// MI455X (gfx1250) — compile-verified, matrix operand order checked by reading, not run on hardware
//
#include <hip/hip_runtime.h>

// ============================================================================
// ConsMax attention for MI455X (gfx1250, wave32, WMMA f32_16x16x32_f16)
//
// ~69 GFLOP vs ~85 MB unique traffic => compute-bound; all matmuls run on the
// WMMA pipe in f16 with f32 accumulation. This revision adds:
//   * double-buffered LDS staging of the (block-shared) weight tile in the
//     projection GEMMs (split global-load / lds-store phases around the WMMAs)
//   * 32x64 output per wave in the GEMMs (8 WMMAs per K-step)
//   * register ping-pong of K fragments + early V fragment issue in the
//     attention main loop (manually 2x-unrolled, no dynamic vector indexing)
// ============================================================================

typedef __attribute__((ext_vector_type(16))) _Float16 v16h;
typedef __attribute__((ext_vector_type(8)))  _Float16 v8h;
typedef __attribute__((ext_vector_type(4)))  _Float16 v4h;
typedef __attribute__((ext_vector_type(8)))  float    v8f;

#define BATCH 2
#define SEQ   2048
#define HIDN  1024
#define NHEAD 16
#define HDIM  64
#define BS    (BATCH * SEQ)   // 4096 rows

// ----------------------------------------------------------------------------
// 16x32 f16 fragment load (A-operand layout; same register image serves as the
// B operand when the source is stored N-major x K-contiguous => A * B^T).
// CDNA5 ISA 7.12.2 (16-bit A, 16x32):
//   lanes 0..15 : row = lane,    K in {0..7, 16..23}
//   lanes 16..31: row = lane-16, K in {8..15, 24..31}
// ----------------------------------------------------------------------------
__device__ __forceinline__ v16h load_frag16(const _Float16* __restrict__ base, int ld) {
  const int lane = threadIdx.x & 31;
  const int row  = lane & 15;
  const int koff = (lane >> 4) << 3;   // 0 or 8
  const _Float16* p = base + (long)row * ld + koff;
  v8h lo = *(const v8h*)(p);           // K = koff .. koff+7
  v8h hi = *(const v8h*)(p + 16);      // K = koff+16 .. koff+23
  return __builtin_shufflevector(lo, hi, 0,1,2,3,4,5,6,7,8,9,10,11,12,13,14,15);
}

__device__ __forceinline__ v8f wmma_f16(v16h a, v16h b, v8f c) {
  return __builtin_amdgcn_wmma_f32_16x16x32_f16(false, a, false, b, (short)0, c,
                                                false, false);
}

// ----------------------------------------------------------------------------
// fp32 -> f16 conversion (vectorized x4), grid-stride
// ----------------------------------------------------------------------------
__global__ void cvt_f32_to_f16_kernel(const float4* __restrict__ src,
                                      _Float16* __restrict__ dst, int n4) {
  int i = blockIdx.x * blockDim.x + threadIdx.x;
  const int stride = gridDim.x * blockDim.x;
  for (; i < n4; i += stride) {
    float4 v = src[i];
    v4h o = { (_Float16)v.x, (_Float16)v.y, (_Float16)v.z, (_Float16)v.w };
    ((v4h*)dst)[i] = o;
  }
}

// ----------------------------------------------------------------------------
// Shared GEMM mainloop: per-block 128(M)x64(N), 4 waves x (32x64 each).
// Weight chunk 64(N)x32(K) f16 double-buffered in LDS; global loads for the
// next chunk are issued BEFORE the WMMA block, the lds-stores after it, so
// memory latency hides behind 8 WMMAs per iteration.
// ----------------------------------------------------------------------------
__device__ __forceinline__ void stage_load(const _Float16* __restrict__ Wb,
                                           int k0, int tid, v8h g[2]) {
  #pragma unroll
  for (int i = 0; i < 2; ++i) {
    const int s = tid + i * 128;     // 256 16-byte segments total
    const int row = s >> 2;          // 0..63
    const int seg = s & 3;           // 16B segment within 64B row
    g[i] = *(const v8h*)(Wb + (long)row * HIDN + k0 + seg * 8);
  }
}

__device__ __forceinline__ void stage_store(_Float16* bTile, int tid, const v8h g[2]) {
  #pragma unroll
  for (int i = 0; i < 2; ++i) {
    const int s = tid + i * 128;
    const int row = s >> 2;
    const int seg = s & 3;
    *(v8h*)(bTile + row * 32 + seg * 8) = g[i];
  }
}

__device__ __forceinline__ void gemm_mainloop(const _Float16* __restrict__ A,
                                              const _Float16* __restrict__ W,
                                              _Float16* bTile,   // [2][64*32] LDS
                                              v8f (&acc)[2][4]) {
  const int tid = threadIdx.x;
  v8h g[2];
  stage_load(W, 0, tid, g);
  stage_store(bTile, tid, g);
  __syncthreads();

  int buf = 0;
  for (int k0 = 0; k0 < HIDN; k0 += 32, buf ^= 1) {
    const int kn = k0 + 32;
    if (kn < HIDN) stage_load(W, kn, tid, g);        // next chunk: global->regs

    v16h a0 = load_frag16(A + k0, HIDN);             // two M-tiles of A
    v16h a1 = load_frag16(A + (long)16 * HIDN + k0, HIDN);

    _Float16* bt = bTile + buf * (64 * 32);
    #pragma unroll
    for (int t = 0; t < 4; ++t) {
      v16h b = load_frag16(bt + t * 16 * 32, 32);    // ds_load from LDS
      acc[0][t] = wmma_f16(a0, b, acc[0][t]);
      acc[1][t] = wmma_f16(a1, b, acc[1][t]);
    }

    if (kn < HIDN) stage_store(bTile + (buf ^ 1) * (64 * 32), tid, g);
    __syncthreads();
  }
}

// ----------------------------------------------------------------------------
// Fused Q/K/V projection. grid = (BS/128, 48); block = 128.
// blockIdx.y: [0,16)=Q, [16,32)=K, [32,48)=V.
// Q pre-scaled by 1/sqrt(HD)=0.125; V written TRANSPOSED [b,h,d,s].
// ----------------------------------------------------------------------------
__global__ __launch_bounds__(128)
void qkv_proj_kernel(const _Float16* __restrict__ hs,
                     const _Float16* __restrict__ wq,
                     const _Float16* __restrict__ wk,
                     const _Float16* __restrict__ wv,
                     const float* __restrict__ bq,
                     const float* __restrict__ bk,
                     const float* __restrict__ bv,
                     _Float16* __restrict__ qOut,
                     _Float16* __restrict__ kOut,
                     _Float16* __restrict__ vtOut) {
  __shared__ alignas(16) _Float16 bTile[2 * 64 * 32];   // 8 KB

  const int wave  = threadIdx.x >> 5;
  const int lane  = threadIdx.x & 31;
  const int srow0 = blockIdx.x * 128 + wave * 32;
  const int proj  = blockIdx.y >> 4;                    // 0=Q 1=K 2=V
  const int ocol0 = (blockIdx.y & 15) * 64;

  const _Float16* __restrict__ A = hs + (long)srow0 * HIDN;
  const _Float16* __restrict__ W =
      (proj == 0 ? wq : proj == 1 ? wk : wv) + (long)ocol0 * HIDN;
  const float* __restrict__ bias = (proj == 0 ? bq : proj == 1 ? bk : bv);

  v8f acc[2][4];
  #pragma unroll
  for (int m = 0; m < 2; ++m)
    #pragma unroll
    for (int t = 0; t < 4; ++t) acc[m][t] = (v8f){};

  gemm_mainloop(A, W, bTile, acc);

  const int hiHalf = lane >> 4;
  const int n      = lane & 15;
  #pragma unroll
  for (int t = 0; t < 4; ++t) {
    const int o = ocol0 + t * 16 + n;
    const int h = o >> 6, d = o & 63;
    const float bval = bias[o];
    #pragma unroll
    for (int m = 0; m < 2; ++m) {
      #pragma unroll
      for (int r = 0; r < 8; ++r) {
        const int sg   = srow0 + m * 16 + r + 8 * hiHalf;   // [0, BS)
        const int bb   = sg >> 11;
        const int sloc = sg & (SEQ - 1);
        const long bh  = (long)bb * NHEAD + h;
        const float y  = acc[m][t][r] + bval;
        if (proj == 0)
          qOut[(bh * SEQ + sloc) * HDIM + d] = (_Float16)(y * 0.125f);
        else if (proj == 1)
          kOut[(bh * SEQ + sloc) * HDIM + d] = (_Float16)y;
        else
          vtOut[(bh * HDIM + d) * SEQ + sloc] = (_Float16)y;
      }
    }
  }
}

// ----------------------------------------------------------------------------
// Streaming ConsMax attention. grid = (SEQ/64, B*NH); block = 128 (4 waves).
// Each wave owns a 16-query tile and streams keys in chunks of 32; the K
// fragments are register ping-pong double-buffered via manual 2x unroll,
// V fragments are issued before the VALU-heavy max/exp phase.
// ----------------------------------------------------------------------------
#define ATTN_PRELOAD(D00, D01, D10, D11, KBASE)                                 \
  D00 = load_frag16(Kp + (long)(KBASE) * HDIM, HDIM);                           \
  D01 = load_frag16(Kp + (long)(KBASE) * HDIM + 32, HDIM);                      \
  D10 = load_frag16(Kp + (long)((KBASE) + 16) * HDIM, HDIM);                    \
  D11 = load_frag16(Kp + (long)((KBASE) + 16) * HDIM + 32, HDIM);

#define ATTN_PROCESS(KC00, KC01, KC10, KC11, K0V)                               \
  {                                                                             \
    /* V fragments for this chunk: issue early, consume last */                 \
    v16h vb0 = load_frag16(Vtp + (long)0  * SEQ + (K0V), SEQ);                  \
    v16h vb1 = load_frag16(Vtp + (long)16 * SEQ + (K0V), SEQ);                  \
    v16h vb2 = load_frag16(Vtp + (long)32 * SEQ + (K0V), SEQ);                  \
    v16h vb3 = load_frag16(Vtp + (long)48 * SEQ + (K0V), SEQ);                  \
    /* scores for 32 keys: two 16x16 C tiles, K-dim 64 */                       \
    v8f c0 = (v8f){}, c1 = (v8f){};                                             \
    c0 = wmma_f16(aq0, KC00, c0);                                               \
    c0 = wmma_f16(aq1, KC01, c0);                                               \
    c1 = wmma_f16(aq0, KC10, c1);                                               \
    c1 = wmma_f16(aq1, KC11, c1);                                               \
    const float madd0 = (1.0f - mrow[(K0V) + n]) * -10000.0f - betav;           \
    const float madd1 = (1.0f - mrow[(K0V) + 16 + n]) * -10000.0f - betav;      \
    _Pragma("unroll") for (int r = 0; r < 8; ++r) {                             \
      c0[r] += madd0; c1[r] += madd1;                                           \
    }                                                                           \
    /* rowmax across the 16 columns (lanes) of each half */                     \
    float tmax[8];                                                              \
    _Pragma("unroll") for (int r = 0; r < 8; ++r) tmax[r] = fmaxf(c0[r], c1[r]);\
    _Pragma("unroll") for (int xm = 1; xm < 16; xm <<= 1) {                     \
      _Pragma("unroll") for (int r = 0; r < 8; ++r)                             \
        tmax[r] = fmaxf(tmax[r], __shfl_xor(tmax[r], xm, 32));                  \
    }                                                                           \
    float scl[8];                                                               \
    _Pragma("unroll") for (int r = 0; r < 8; ++r) {                             \
      const float mn = fmaxf(mrun[r], tmax[r]);                                 \
      scl[r]  = __expf(mrun[r] - mn);                                           \
      mrun[r] = mn;                                                             \
    }                                                                           \
    _Pragma("unroll") for (int t = 0; t < 4; ++t)                               \
      _Pragma("unroll") for (int r = 0; r < 8; ++r) o[t][r] *= scl[r];          \
    /* P = exp(S - m): C-layout -> row-major f16 via per-wave LDS */            \
    _Pragma("unroll") for (int r = 0; r < 8; ++r) {                             \
      my[(r + 8 * hiHalf) * 32 + n]      = (_Float16)__expf(c0[r] - mrun[r]);   \
      my[(r + 8 * hiHalf) * 32 + 16 + n] = (_Float16)__expf(c1[r] - mrun[r]);   \
    }                                                                           \
    v16h ap = load_frag16(my, 32);                                              \
    o[0] = wmma_f16(ap, vb0, o[0]);                                             \
    o[1] = wmma_f16(ap, vb1, o[1]);                                             \
    o[2] = wmma_f16(ap, vb2, o[2]);                                             \
    o[3] = wmma_f16(ap, vb3, o[3]);                                             \
  }

__global__ __launch_bounds__(128)
void consmax_attn_kernel(const _Float16* __restrict__ q,
                         const _Float16* __restrict__ kmat,
                         const _Float16* __restrict__ vt,
                         const float* __restrict__ amask,
                         const float* __restrict__ betaP,
                         const float* __restrict__ gammaP,
                         _Float16* __restrict__ ctx) {
  __shared__ alignas(16) _Float16 pbuf[4][16 * 32];   // per-wave P staging

  const int wave = threadIdx.x >> 5;
  const int lane = threadIdx.x & 31;
  const int bh   = blockIdx.y;
  const int b    = bh >> 4, h = bh & 15;
  const int s0   = blockIdx.x * 64 + wave * 16;

  const _Float16* __restrict__ Qp  = q    + ((long)bh * SEQ + s0) * HDIM;
  const _Float16* __restrict__ Kp  = kmat + (long)bh * SEQ * HDIM;
  const _Float16* __restrict__ Vtp = vt   + (long)bh * HDIM * SEQ;
  const float* __restrict__ mrow   = amask + (long)b * SEQ;
  const float betav    = betaP[0];
  const float invGamma = 1.0f / gammaP[0];

  const v16h aq0 = load_frag16(Qp,      HDIM);   // Q tile 16x64 -> two A frags
  const v16h aq1 = load_frag16(Qp + 32, HDIM);

  v8f o[4];
  #pragma unroll
  for (int t = 0; t < 4; ++t) o[t] = (v8f){};
  float mrun[8];
  #pragma unroll
  for (int r = 0; r < 8; ++r) mrun[r] = -1e30f;

  const int hiHalf = lane >> 4;
  const int n      = lane & 15;
  _Float16* __restrict__ my = pbuf[wave];

  // Register ping-pong of K fragments; two 32-key chunks per trip.
  v16h x00, x01, x10, x11, y00, y01, y10, y11;
  ATTN_PRELOAD(x00, x01, x10, x11, 0)
  for (int k0 = 0; k0 < SEQ; k0 += 64) {
    ATTN_PRELOAD(y00, y01, y10, y11, k0 + 32)
    ATTN_PROCESS(x00, x01, x10, x11, k0)
    if (k0 + 64 < SEQ) {
      ATTN_PRELOAD(x00, x01, x10, x11, k0 + 64)
    }
    ATTN_PROCESS(y00, y01, y10, y11, k0 + 32)
  }

  // ctx[b, s, h*64+d] = O / gamma  (f16)
  #pragma unroll
  for (int t = 0; t < 4; ++t)
    #pragma unroll
    for (int r = 0; r < 8; ++r) {
      const long row = (long)b * SEQ + s0 + r + 8 * hiHalf;
      ctx[row * HIDN + h * HDIM + t * 16 + n] = (_Float16)(o[t][r] * invGamma);
    }
}

// ----------------------------------------------------------------------------
// Output projection: out = ctx * Wo^T + bo (fp32). grid = (BS/128, HIDN/64).
// ----------------------------------------------------------------------------
__global__ __launch_bounds__(128)
void out_proj_kernel(const _Float16* __restrict__ ctx,
                     const _Float16* __restrict__ wo,
                     const float* __restrict__ bo,
                     float* __restrict__ out) {
  __shared__ alignas(16) _Float16 bTile[2 * 64 * 32];   // 8 KB

  const int wave  = threadIdx.x >> 5;
  const int lane  = threadIdx.x & 31;
  const int srow0 = blockIdx.x * 128 + wave * 32;
  const int ocol0 = blockIdx.y * 64;

  const _Float16* __restrict__ A = ctx + (long)srow0 * HIDN;
  const _Float16* __restrict__ W = wo + (long)ocol0 * HIDN;

  v8f acc[2][4];
  #pragma unroll
  for (int m = 0; m < 2; ++m)
    #pragma unroll
    for (int t = 0; t < 4; ++t) acc[m][t] = (v8f){};

  gemm_mainloop(A, W, bTile, acc);

  const int hiHalf = lane >> 4;
  const int n      = lane & 15;
  #pragma unroll
  for (int t = 0; t < 4; ++t) {
    const int o = ocol0 + t * 16 + n;
    const float bval = bo[o];
    #pragma unroll
    for (int m = 0; m < 2; ++m) {
      #pragma unroll
      for (int r = 0; r < 8; ++r) {
        const long row = srow0 + m * 16 + r + 8 * hiHalf;
        out[row * HIDN + o] = acc[m][t][r] + bval;
      }
    }
  }
}

// ----------------------------------------------------------------------------
// Launch
// ----------------------------------------------------------------------------
extern "C" void kernel_launch(void* const* d_in, const int* in_sizes, int n_in,
                              void* d_out, int out_size, void* d_ws, size_t ws_size,
                              hipStream_t stream) {
  (void)in_sizes; (void)n_in; (void)out_size; (void)ws_size;

  const float* hs    = (const float*)d_in[0];
  const float* amask = (const float*)d_in[1];
  const float* Wq    = (const float*)d_in[2];
  const float* bq    = (const float*)d_in[3];
  const float* Wk    = (const float*)d_in[4];
  const float* bk    = (const float*)d_in[5];
  const float* Wv    = (const float*)d_in[6];
  const float* bv    = (const float*)d_in[7];
  const float* Wo    = (const float*)d_in[8];
  const float* bo    = (const float*)d_in[9];
  const float* beta  = (const float*)d_in[10];
  const float* gamma = (const float*)d_in[11];

  const size_t HS = (size_t)BS * HIDN;    // 4,194,304 halves
  const size_t WE = (size_t)HIDN * HIDN;  // 1,048,576 halves

  // f16 workspace layout (total 25,165,824 halves = 48 MB)
  _Float16* hsF  = (_Float16*)d_ws;
  _Float16* wqF  = hsF + HS;
  _Float16* wkF  = wqF + WE;
  _Float16* wvF  = wkF + WE;
  _Float16* woF  = wvF + WE;
  _Float16* Qb   = woF + WE;
  _Float16* Kb   = Qb + HS;
  _Float16* Vtb  = Kb + HS;
  _Float16* ctxF = Vtb + HS;

  // 1) fp32 -> f16 conversions (bandwidth-trivial)
  cvt_f32_to_f16_kernel<<<2048, 256, 0, stream>>>((const float4*)hs, hsF, (int)(HS / 4));
  cvt_f32_to_f16_kernel<<<1024, 256, 0, stream>>>((const float4*)Wq, wqF, (int)(WE / 4));
  cvt_f32_to_f16_kernel<<<1024, 256, 0, stream>>>((const float4*)Wk, wkF, (int)(WE / 4));
  cvt_f32_to_f16_kernel<<<1024, 256, 0, stream>>>((const float4*)Wv, wvF, (int)(WE / 4));
  cvt_f32_to_f16_kernel<<<1024, 256, 0, stream>>>((const float4*)Wo, woF, (int)(WE / 4));

  // 2) fused QKV projection (Q scaled by 1/sqrt(HD), V stored transposed)
  qkv_proj_kernel<<<dim3(BS / 128, 48), dim3(128), 0, stream>>>(
      hsF, wqF, wkF, wvF, bq, bk, bv, Qb, Kb, Vtb);

  // 3) streaming ConsMax attention
  consmax_attn_kernel<<<dim3(SEQ / 64, BATCH * NHEAD), dim3(128), 0, stream>>>(
      Qb, Kb, Vtb, amask, beta, gamma, ctxF);

  // 4) output projection -> fp32
  out_proj_kernel<<<dim3(BS / 128, HIDN / 64), dim3(128), 0, stream>>>(
      ctxF, woF, bo, (float*)d_out);
}
